// M2GNN_word_62277025792393
// MI455X (gfx1250) — matrix operand order repaired and checked by reading, not compiled, not run on hardware
//
#include <hip/hip_runtime.h>
#include <math.h>

#define DIM   128
#define NHEAD 8
#define NB    32768
#define NUQ   32768
#define NTQ   16384
#define NE    262144

typedef float    v2f  __attribute__((ext_vector_type(2)));
typedef float    v8f  __attribute__((ext_vector_type(8)));
typedef _Float16 v16h __attribute__((ext_vector_type(16)));

// ---------------------------------------------------------------- utilities

__device__ __forceinline__ float wred32(float v) {
  #pragma unroll
  for (int m = 16; m >= 1; m >>= 1) v += __shfl_xor(v, m, 32);
  return v;
}

__device__ __forceinline__ float sigmoidf_(float x) { return 1.f / (1.f + __expf(-x)); }

__device__ __forceinline__ void atomicMaxF(float* addr, float val) {
  // signed-max for positives, unsigned-min for negatives; correct with -inf init
  if (val >= 0.f) atomicMax((int*)addr, __float_as_int(val));
  else            atomicMin((unsigned int*)addr, (unsigned int)__float_as_int(val));
}

// One wave32 computes a 16x16 f32 tile of  C = A(16xK) * W(Kx128)[:, n0:n0+16].
// A is in LDS (row stride K). Uses V_WMMA_F32_16X16X4_F32 (full fp32) when the
// builtin exists; otherwise falls back to the confirmed f16 16x16x32 WMMA.
__device__ __forceinline__ void wmma_tile(const float* As, const float* W,
                                          int K, int n0, float* cOut) {
  const int lane = threadIdx.x & 31;
  const int hi = lane >> 4, l15 = lane & 15;
#if __has_builtin(__builtin_amdgcn_wmma_f32_16x16x4_f32)
  v8f c = {};
  for (int k = 0; k < K; k += 4) {
    v2f a, b;
    a[0] = As[l15 * K + k + 2 * hi];
    a[1] = As[l15 * K + k + 2 * hi + 1];
    b[0] = W[(k + 2 * hi) * DIM + n0 + l15];
    b[1] = W[(k + 2 * hi + 1) * DIM + n0 + l15];
    c = __builtin_amdgcn_wmma_f32_16x16x4_f32(false, a, false, b, (short)0, c, false, false);
  }
#else
  v8f c = {};
  for (int k0 = 0; k0 < K; k0 += 32) {
    v16h a, b;
    #pragma unroll
    for (int vv = 0; vv < 8; vv++) {
      int ka = hi * 16 + 2 * vv;
      a[2 * vv]     = (_Float16)As[l15 * K + k0 + ka];
      a[2 * vv + 1] = (_Float16)As[l15 * K + k0 + ka + 1];
      int kb = hi * 16 + 2 * vv;
      b[2 * vv]     = (_Float16)W[(k0 + kb) * DIM + n0 + l15];
      b[2 * vv + 1] = (_Float16)W[(k0 + kb + 1) * DIM + n0 + l15];
    }
    c = __builtin_amdgcn_wmma_f32_16x16x32_f16(false, a, false, b, (short)0, c, false, false);
  }
#endif
  #pragma unroll
  for (int j = 0; j < 8; j++) cOut[j] = c[j];
}

// ---------------------------------------------------------------- kernels

__global__ void k_fill(float* p, float v, long long n) {
  long long i = (long long)blockIdx.x * blockDim.x + threadIdx.x;
  if (i < n) p[i] = v;
}

// Generic Y[N x 128] = X[N x K<=128] @ W[K x 128] (+bias)(+relu); 1 wave / 16x16 tile.
__global__ void __launch_bounds__(32)
k_gemm_nn(const float* __restrict__ X, const float* __restrict__ W,
          float* __restrict__ Y, const float* __restrict__ bias,
          int Nrows, int K, int ldY, int dorelu) {
  __shared__ float xs[16 * 128];
  const int lane = threadIdx.x, hi = lane >> 4, l15 = lane & 15;
  const int r0 = blockIdx.x * 16, n0 = blockIdx.y * 16;
  for (int idx = lane; idx < 16 * K; idx += 32) {
    int rr = idx / K, kk = idx - rr * K;
    int row = r0 + rr;
    xs[idx] = (row < Nrows) ? X[(size_t)row * K + kk] : 0.f;
  }
  __syncthreads();
  float c[8];
  wmma_tile(xs, W, K, n0, c);
  float bv = bias ? bias[n0 + l15] : 0.f;
  #pragma unroll
  for (int j = 0; j < 8; j++) {
    int row = r0 + j + 8 * hi;
    if (row < Nrows) {
      float v = c[j] + bv;
      if (dorelu) v = v > 0.f ? v : 0.f;
      Y[(size_t)row * ldY + n0 + l15] = v;
    }
  }
}

// Gated low-level user features -> node_feats[0].  1 wave handles 16 users.
__global__ void __launch_bounds__(32)
k_user_low(const float* __restrict__ ue, const float* __restrict__ fl_w,
           const float* __restrict__ fl_b, const float* __restrict__ gw1,
           const float* __restrict__ gb1, const float* __restrict__ gw2,
           const float* __restrict__ gb2, float* __restrict__ nf) {
  __shared__ float sf[16];
  __shared__ float femb[16 * 128];
  __shared__ float gbuf[16];
  __shared__ float lowacc[16 * 128];
  const int lane = threadIdx.x, hi = lane >> 4, l15 = lane & 15;
  const int u0 = blockIdx.x * 16;
  for (int idx = lane; idx < 2048; idx += 32) lowacc[idx] = 0.f;
  for (int i = 0; i < 8; i++) {
    if (lane < 16) sf[lane] = ue[(size_t)(u0 + lane) * DIM + i];
    __syncthreads();
    for (int idx = lane; idx < 2048; idx += 32) {
      int u = idx >> 7, d = idx & 127;
      femb[idx] = sf[u] * fl_w[i * 128 + d] + fl_b[i * 128 + d];
    }
    __syncthreads();
    float dotacc[8];
    #pragma unroll
    for (int j = 0; j < 8; j++) dotacc[j] = 0.f;
    for (int n0 = 0; n0 < 128; n0 += 16) {
      float c[8];
      wmma_tile(femb, gw1 + (size_t)i * 128 * 128, 128, n0, c);
      float w2v = gw2[i * 128 + n0 + l15];
      float bv  = gb1[i * 128 + n0 + l15];
      #pragma unroll
      for (int j = 0; j < 8; j++) {
        float h = c[j] + bv;
        h = h > 0.f ? h : 0.f;
        dotacc[j] += h * w2v;          // dot with gw2 over output dim
      }
    }
    #pragma unroll
    for (int j = 0; j < 8; j++)
      for (int msk = 1; msk < 16; msk <<= 1)
        dotacc[j] += __shfl_xor(dotacc[j], msk, 32);
    if (l15 == 0) {
      #pragma unroll
      for (int j = 0; j < 8; j++)
        gbuf[hi * 8 + j] = sigmoidf_(dotacc[j] + gb2[i]);
    }
    __syncthreads();
    for (int idx = lane; idx < 2048; idx += 32)
      lowacc[idx] += gbuf[idx >> 7] * femb[idx];
    __syncthreads();
  }
  for (int idx = lane; idx < 2048; idx += 32) {
    int u = idx >> 7, d = idx & 127;
    nf[(size_t)(u0 + u) * DIM + d] = lowacc[idx] * 0.125f;  // mean over 8 features
  }
}

// el/er: per (node, head) 16-wide dot with attention vector
__global__ void k_elr(const float* __restrict__ proj, const float* __restrict__ avec,
                      float* __restrict__ out, int N) {
  int t = blockIdx.x * blockDim.x + threadIdx.x;
  if (t >= N * NHEAD) return;
  int n = t >> 3, h = t & 7;
  float a = 0.f;
  #pragma unroll
  for (int k = 0; k < 16; k++) a += proj[(size_t)n * DIM + h * 16 + k] * avec[h * 16 + k];
  out[t] = a;
}

__global__ void k_edge_max(const int* __restrict__ src, const int* __restrict__ dst,
                           const float* __restrict__ el, const float* __restrict__ er,
                           float* __restrict__ m, float* __restrict__ ebuf) {
  int t = blockIdx.x * blockDim.x + threadIdx.x;
  if (t >= NE * NHEAD) return;
  int e = t >> 3, h = t & 7;
  int s = src[e], d = dst[e];
  float x = el[s * NHEAD + h] + er[d * NHEAD + h];
  x = x > 0.f ? x : 0.2f * x;          // leaky relu
  ebuf[t] = x;
  atomicMaxF(&m[d * NHEAD + h], x);
}

__global__ void k_edge_exp(const int* __restrict__ dst, const float* __restrict__ m,
                           float* __restrict__ ebuf, float* __restrict__ den) {
  int t = blockIdx.x * blockDim.x + threadIdx.x;
  if (t >= NE * NHEAD) return;
  int e = t >> 3, h = t & 7;
  int d = dst[e];
  float mv = m[d * NHEAD + h];
  if (!(mv > -3.0e38f)) mv = 0.f;      // where(isfinite(m), m, 0)
  float ex = __expf(ebuf[t] - mv);
  ebuf[t] = ex;
  atomicAdd(&den[d * NHEAD + h], ex);
}

__global__ void k_edge_scatter(const int* __restrict__ src, const int* __restrict__ dst,
                               const float* __restrict__ fs, const float* __restrict__ ebuf,
                               const float* __restrict__ den, float* __restrict__ agg) {
  long long t = (long long)blockIdx.x * blockDim.x + threadIdx.x;
  if (t >= (long long)NE * DIM) return;
  int e = (int)(t >> 7), c = (int)(t & 127), h = c >> 4;
  int s = src[e], d = dst[e];
  float alpha = ebuf[e * NHEAD + h] / (den[d * NHEAD + h] + 1e-9f);
  atomicAdd(&agg[(size_t)d * DIM + c], alpha * fs[(size_t)s * DIM + c]);
}

__global__ void k_bias_add(float* __restrict__ agg, const float* __restrict__ bias, int N) {
  int t = blockIdx.x * blockDim.x + threadIdx.x;
  if (t < N * DIM) agg[t] += bias[t & 127];
}

__global__ void k_probs(const float* __restrict__ h, const float* __restrict__ w2,
                        const float* __restrict__ b2, float* __restrict__ probs, int N) {
  int b = blockIdx.x * blockDim.x + threadIdx.x;
  if (b >= N) return;
  float a0 = b2[0], a1 = b2[1];
  for (int d = 0; d < DIM; d++) {
    float hv = h[(size_t)b * DIM + d];
    a0 += hv * w2[d * 2 + 0];
    a1 += hv * w2[d * 2 + 1];
  }
  probs[b * 2 + 0] = sigmoidf_(a0);
  probs[b * 2 + 1] = sigmoidf_(a1);
}

// Routing 1 (n collapses: broadcast caps): in-place uh -> v1. 1 wave / user.
__global__ void __launch_bounds__(32)
k_route1(const float* __restrict__ uh, float* __restrict__ v1) {
  const int lane = threadIdx.x;
  const size_t b = blockIdx.x;
  float u[4][4], v[4][4];
  #pragma unroll
  for (int m = 0; m < 4; m++)
    #pragma unroll
    for (int j = 0; j < 4; j++)
      u[m][j] = uh[b * 512 + m * 128 + lane * 4 + j];
  float bl[4] = {0.f, 0.f, 0.f, 0.f};
  for (int it = 0; it < 3; it++) {
    float mx = fmaxf(fmaxf(bl[0], bl[1]), fmaxf(bl[2], bl[3]));
    float cs[4], csum = 0.f;
    #pragma unroll
    for (int m = 0; m < 4; m++) { cs[m] = __expf(bl[m] - mx); csum += cs[m]; }
    #pragma unroll
    for (int m = 0; m < 4; m++) {
      float c8 = 8.f * cs[m] / csum;       // sum over n of c*u_hat = 8*c*uh
      float sv[4], sloc = 0.f;
      #pragma unroll
      for (int j = 0; j < 4; j++) { sv[j] = c8 * u[m][j]; sloc += sv[j] * sv[j]; }
      float sn = wred32(sloc);
      float scale = sn / ((1.f + sn) * (sqrtf(sn) + 1e-8f));
      #pragma unroll
      for (int j = 0; j < 4; j++) v[m][j] = sv[j] * scale;
      if (it < 2) {
        float d = 0.f;
        #pragma unroll
        for (int j = 0; j < 4; j++) d += u[m][j] * v[m][j];
        bl[m] += wred32(d);
      }
    }
  }
  #pragma unroll
  for (int m = 0; m < 4; m++)
    #pragma unroll
    for (int j = 0; j < 4; j++)
      v1[b * 512 + m * 128 + lane * 4 + j] = v[m][j];
}

// mid-GAT over the 4 capsules. 1 wave / user, f staged in LDS.
__global__ void __launch_bounds__(32)
k_midgat(const float* __restrict__ v1, const float* __restrict__ w2,
         const float* __restrict__ al2, const float* __restrict__ ar2,
         const float* __restrict__ b2, float* __restrict__ v1g) {
  __shared__ float vrow[512];
  __shared__ float f[4096];
  __shared__ float el[32], er[32];
  __shared__ float alpha[128];            // [s][d][h] -> s*32 + d*8 + h
  const int lane = threadIdx.x;
  const size_t b = blockIdx.x;
  for (int i = lane; i < 512; i += 32) vrow[i] = v1[b * 512 + i];
  __syncthreads();
  for (int idx = lane; idx < 4096; idx += 32) {
    int s = idx >> 10, n = idx & 1023;
    float acc = 0.f;
    for (int k = 0; k < 128; k++) acc += vrow[s * 128 + k] * w2[k * 1024 + n];
    f[idx] = acc;
  }
  __syncthreads();
  {
    int s = lane >> 3, h = lane & 7;
    float a = 0.f, bb = 0.f;
    for (int d = 0; d < 128; d++) {
      float fv = f[s * 1024 + h * 128 + d];
      a  += fv * al2[h * 128 + d];
      bb += fv * ar2[h * 128 + d];
    }
    el[lane] = a; er[lane] = bb;
  }
  __syncthreads();
  {
    int d = lane >> 3, h = lane & 7;
    float es[4], mx = -1e30f;
    #pragma unroll
    for (int s = 0; s < 4; s++) {
      float x;
      if (s == d) x = -1e30f;
      else { x = el[s * 8 + h] + er[d * 8 + h]; x = x > 0.f ? x : 0.2f * x; }
      es[s] = x; if (x > mx) mx = x;
    }
    float dn = 0.f;
    #pragma unroll
    for (int s = 0; s < 4; s++) { es[s] = __expf(es[s] - mx); dn += es[s]; }
    #pragma unroll
    for (int s = 0; s < 4; s++) alpha[s * 32 + d * 8 + h] = es[s] / dn;
  }
  __syncthreads();
  for (int idx = lane; idx < 512; idx += 32) {
    int d = idx >> 7, k = idx & 127;
    float acc = 0.f;
    #pragma unroll
    for (int h = 0; h < 8; h++) {
      float s_ = 0.f;
      #pragma unroll
      for (int s = 0; s < 4; s++) s_ += alpha[s * 32 + d * 8 + h] * f[s * 1024 + h * 128 + k];
      acc += s_ + b2[h * 128 + k];
    }
    v1g[b * 512 + d * 128 + k] = acc * 0.125f;   // mean over heads
  }
}

// Routing 2 (n=4, m=2), u_hat computed on the fly. 1 wave / user.
__global__ void __launch_bounds__(32)
k_route2(const float* __restrict__ v1g, const float* __restrict__ cw2,
         float* __restrict__ v2b) {
  __shared__ float g[512];
  const int lane = threadIdx.x;
  const size_t b = blockIdx.x;
  for (int i = lane; i < 512; i += 32) g[i] = v1g[b * 512 + i];
  __syncthreads();
  float u[4][2][4];
  for (int n = 0; n < 4; n++)
    for (int m = 0; m < 2; m++)
      #pragma unroll
      for (int j = 0; j < 4; j++) {
        float acc = 0.f;
        for (int k = 0; k < 128; k++)
          acc += g[n * 128 + k] * cw2[(size_t)m * 16384 + k * 128 + lane * 4 + j];
        u[n][m][j] = acc;
      }
  float bl[4][2] = {};
  float v[2][4];
  for (int it = 0; it < 3; it++) {
    float c[4][2];
    #pragma unroll
    for (int n = 0; n < 4; n++) {
      float mx = fmaxf(bl[n][0], bl[n][1]);
      float e0 = __expf(bl[n][0] - mx), e1 = __expf(bl[n][1] - mx);
      float s = e0 + e1;
      c[n][0] = e0 / s; c[n][1] = e1 / s;
    }
    #pragma unroll
    for (int m = 0; m < 2; m++) {
      float sv[4], sloc = 0.f;
      #pragma unroll
      for (int j = 0; j < 4; j++) {
        float s_ = 0.f;
        #pragma unroll
        for (int n = 0; n < 4; n++) s_ += c[n][m] * u[n][m][j];
        sv[j] = s_; sloc += s_ * s_;
      }
      float sn = wred32(sloc);
      float scale = sn / ((1.f + sn) * (sqrtf(sn) + 1e-8f));
      #pragma unroll
      for (int j = 0; j < 4; j++) v[m][j] = sv[j] * scale;
    }
    if (it < 2)
      #pragma unroll
      for (int n = 0; n < 4; n++)
        #pragma unroll
        for (int m = 0; m < 2; m++) {
          float d = 0.f;
          #pragma unroll
          for (int j = 0; j < 4; j++) d += u[n][m][j] * v[m][j];
          bl[n][m] += wred32(d);
        }
  }
  #pragma unroll
  for (int m = 0; m < 2; m++)
    #pragma unroll
    for (int j = 0; j < 4; j++)
      v2b[b * 256 + m * 128 + lane * 4 + j] = v[m][j];
}

// Final projection: out = (v2 * probs).reshape(B,256) @ op_w + op_b.  WMMA K=256.
__global__ void __launch_bounds__(32)
k_out(const float* __restrict__ v2b, const float* __restrict__ probs,
      const float* __restrict__ opw, const float* __restrict__ opb,
      float* __restrict__ out, int N) {
  __shared__ float xs[16 * 256];
  const int lane = threadIdx.x, hi = lane >> 4, l15 = lane & 15;
  const int r0 = blockIdx.x * 16, n0 = blockIdx.y * 16;
  for (int idx = lane; idx < 16 * 256; idx += 32) {
    int rr = idx >> 8, kk = idx & 255;
    int row = r0 + rr;
    xs[idx] = v2b[(size_t)row * 256 + kk] * probs[row * 2 + (kk >> 7)];
  }
  __syncthreads();
  float c[8];
  wmma_tile(xs, opw, 256, n0, c);
  float bv = opb[n0 + l15];
  #pragma unroll
  for (int j = 0; j < 8; j++) {
    int row = r0 + j + 8 * hi;
    if (row < N) out[(size_t)row * DIM + n0 + l15] = c[j] + bv;
  }
}

// ---------------------------------------------------------------- host

extern "C" void kernel_launch(void* const* d_in, const int* in_sizes, int n_in,
                              void* d_out, int out_size, void* d_ws, size_t ws_size,
                              hipStream_t stream) {
  (void)in_sizes; (void)n_in; (void)out_size; (void)ws_size;
  const float* user_embed   = (const float*)d_in[0];
  const float* review_embed = (const float*)d_in[1];
  const float* tag_embed    = (const float*)d_in[2];
  const float* fl_w  = (const float*)d_in[3];
  const float* fl_b  = (const float*)d_in[4];
  const float* gw1   = (const float*)d_in[5];
  const float* gb1   = (const float*)d_in[6];
  const float* gw2   = (const float*)d_in[7];
  const float* gb2   = (const float*)d_in[8];
  const float* gat_w = (const float*)d_in[9];
  const float* gat_al= (const float*)d_in[10];
  const float* gat_ar= (const float*)d_in[11];
  const float* gat_b = (const float*)d_in[12];
  const float* w2    = (const float*)d_in[13];
  const float* al2   = (const float*)d_in[14];
  const float* ar2   = (const float*)d_in[15];
  const float* b2    = (const float*)d_in[16];
  const float* cw1   = (const float*)d_in[17];
  const float* cw2   = (const float*)d_in[18];
  const float* gn_w1 = (const float*)d_in[19];
  const float* gn_b1 = (const float*)d_in[20];
  const float* gn_w2 = (const float*)d_in[21];
  const float* gn_b2 = (const float*)d_in[22];
  const float* op_w  = (const float*)d_in[23];
  const float* op_b  = (const float*)d_in[24];
  const int*   edges = (const int*)d_in[25];
  float* out = (float*)d_out;
  float* ws  = (float*)d_ws;

  // workspace layout (floats); total ~66M floats (~266 MB)
  size_t off = 0;
  float* nf_user = ws + off; off += (size_t)NB * DIM;
  float* agg_user= ws + off; off += (size_t)NB * DIM;
  float* projS   = ws + off; off += (size_t)32768 * DIM;
  float* projD   = ws + off; off += (size_t)32768 * DIM;
  float* elb     = ws + off; off += (size_t)32768 * NHEAD;
  float* erb     = ws + off; off += (size_t)32768 * NHEAD;
  float* mb      = ws + off; off += (size_t)32768 * NHEAD;
  float* denb    = ws + off; off += (size_t)32768 * NHEAD;
  float* ebuf    = ws + off; off += (size_t)NE * NHEAD;
  float* gnh     = ws + off; off += (size_t)NB * DIM;
  float* probs   = ws + off; off += (size_t)NB * 2;
  float* uh1     = ws + off; off += (size_t)NB * 4 * DIM;   // also v1 (in-place routing)
  float* v1gb    = ws + off; off += (size_t)NB * 4 * DIM;
  float* v2b     = ws + off; off += (size_t)NB * 2 * DIM;

  float* aggT[3] = { agg_user, out + (size_t)NB * DIM, out + (size_t)NB * DIM + (size_t)NUQ * DIM };
  const float* feat[3] = { nf_user, review_embed, tag_embed };
  const int cnt[3] = { NB, NUQ, NTQ };
  const int relSt[6] = {2, 1, 2, 1, 0, 2};
  const int relDt[6] = {0, 0, 2, 2, 1, 1};

  #define FILL(p, v, n) k_fill<<<(unsigned)(((n) + 255) / 256), 256, 0, stream>>>((p), (v), (long long)(n))

  FILL(agg_user, 0.f, (size_t)NB * DIM);
  FILL(aggT[1], 0.f, (size_t)NUQ * DIM);
  FILL(aggT[2], 0.f, (size_t)NTQ * DIM);

  // node_feats[0]: gated low-level user features (WMMA)
  k_user_low<<<NB / 16, 32, 0, stream>>>(user_embed, fl_w, fl_b, gw1, gb1, gw2, gb2, nf_user);

  for (int r = 0; r < 6; r++) {
    int st = relSt[r], dt = relDt[r];
    const float* WW = gat_w + (size_t)r * DIM * DIM;
    k_gemm_nn<<<dim3(cnt[st] / 16, 8), 32, 0, stream>>>(feat[st], WW, projS, nullptr, cnt[st], DIM, DIM, 0);
    const float* fsP = projS;
    const float* fdP = projS;
    if (st != dt) {
      k_gemm_nn<<<dim3(cnt[dt] / 16, 8), 32, 0, stream>>>(feat[dt], WW, projD, nullptr, cnt[dt], DIM, DIM, 0);
      fdP = projD;
    }
    k_elr<<<(cnt[st] * NHEAD + 255) / 256, 256, 0, stream>>>(fsP, gat_al + r * 128, elb, cnt[st]);
    k_elr<<<(cnt[dt] * NHEAD + 255) / 256, 256, 0, stream>>>(fdP, gat_ar + r * 128, erb, cnt[dt]);
    FILL(mb, -__builtin_inff(), (size_t)cnt[dt] * NHEAD);
    FILL(denb, 0.f, (size_t)cnt[dt] * NHEAD);
    const int* srcI = edges + (size_t)r * 2 * NE;
    const int* dstI = srcI + NE;
    k_edge_max<<<(NE * NHEAD + 255) / 256, 256, 0, stream>>>(srcI, dstI, elb, erb, mb, ebuf);
    k_edge_exp<<<(NE * NHEAD + 255) / 256, 256, 0, stream>>>(dstI, mb, ebuf, denb);
    k_edge_scatter<<<(unsigned)(((long long)NE * DIM + 255) / 256), 256, 0, stream>>>(srcI, dstI, fsP, ebuf, denb, aggT[dt]);
    k_bias_add<<<(cnt[dt] * DIM + 255) / 256, 256, 0, stream>>>(aggT[dt], gat_b + r * 128, cnt[dt]);
  }

  // gating MLP -> probs
  k_gemm_nn<<<dim3(NB / 16, 8), 32, 0, stream>>>(agg_user, gn_w1, gnh, gn_b1, NB, DIM, DIM, 1);
  k_probs<<<(NB + 255) / 256, 256, 0, stream>>>(gnh, gn_w2, gn_b2, probs, NB);

  // capsule transform 1 (broadcast caps => 4 GEMMs) + collapsed routing
  for (int m = 0; m < 4; m++)
    k_gemm_nn<<<dim3(NB / 16, 8), 32, 0, stream>>>(agg_user, cw1 + (size_t)m * DIM * DIM,
                                                   uh1 + m * DIM, nullptr, NB, DIM, 4 * DIM, 0);
  k_route1<<<NB, 32, 0, stream>>>(uh1, uh1);
  k_midgat<<<NB, 32, 0, stream>>>(uh1, w2, al2, ar2, b2, v1gb);
  k_route2<<<NB, 32, 0, stream>>>(v1gb, cw2, v2b);
  k_out<<<dim3(NB / 16, 8), 32, 0, stream>>>(v2b, probs, op_w, op_b, out, NB);

  #undef FILL
}
